// DGMNet_76484777607674
// MI455X (gfx1250) — matrix operand order repaired and checked by reading, not compiled, not run on hardware
//
#include <hip/hip_runtime.h>
#include <hip/hip_bf16.h>
#include <cstdint>

// ---------------------------------------------------------------------------
// DGM net fused kernel for gfx1250 (MI455X), v3.
//
// Roofline: 0.82 TFLOP of GEMM (~1us at matrix-core rate); binding resource is
// L2 bandwidth for weight re-reads (2048 WGs x 6MB packed weights). Blocking:
// waves partition N, each wave computes all 4 row strips, so each B fragment
// feeds 4 WMMAs (global:wmma = 1:2). H/F/T resident in 194KB of the 320KB WGP
// LDS; all 12 512x512 GEMMs on v_wmma_f32_16x16x32_bf16.
//
// v3: activations use v_rcp_f32 (__builtin_amdgcn_rcpf) instead of precise
// IEEE division -- kills the v_div_scale/div_fmas chains that dominated VALU
// count and pushed register allocation into the VGPR-MSB range.
// ---------------------------------------------------------------------------

typedef __attribute__((ext_vector_type(16))) __bf16        v16bf;
typedef __attribute__((ext_vector_type(8)))  float         v8f;
typedef __attribute__((ext_vector_type(8)))  unsigned int  v8u;

// LDS layout (dwords)
constexpr int H_DW   = 0;       // H tile, 64x512 bf16  (16384 dwords)
constexpr int T_DW   = 16384;   // sig(U), then T=sig(U)*tanh(O1), bf16
constexpr int F_DW   = 32768;   // F gate, bf16
constexpr int XS_DW  = 49152;   // X tile, 64x4 f32     (256 dwords)
constexpr int RED_DW = 49408;   // reduction scratch    (256 dwords)
constexpr int SMEM_DW = 49664;
constexpr size_t SMEM_BYTES = (size_t)SMEM_DW * 4;   // 198656 B < 320KB WGP LDS

constexpr int T_E = 32768;   // u16-element bases
constexpr int F_E = 65536;

constexpr int MAT_DW = 131072;   // 512*512 bf16 = 131072 packed dwords / matrix

__device__ __forceinline__ uint32_t bf16_rne(float f) {
    uint32_t u = __float_as_uint(f);
    return (u + 0x7FFFu + ((u >> 16) & 1u)) >> 16;
}
__device__ __forceinline__ float bf16_to_f(uint32_t h) {
    return __uint_as_float(h << 16);
}
// Fast activations: v_exp_f32 + v_rcp_f32, no IEEE divide.
__device__ __forceinline__ float frcp_fast(float x) { return __builtin_amdgcn_rcpf(x); }
__device__ __forceinline__ float fsigmoid(float x)  { return frcp_fast(1.0f + __expf(-x)); }
__device__ __forceinline__ float fsilu(float x)     { return x * fsigmoid(x); }
__device__ __forceinline__ float ftanh_(float x) {
    return fmaf(2.0f, frcp_fast(1.0f + __expf(-2.0f * x)), -1.0f);
}

__device__ __forceinline__ v16bf make_frag(const uint4& a, const uint4& b) {
    v8u t;
    t[0] = a.x; t[1] = a.y; t[2] = a.z; t[3] = a.w;
    t[4] = b.x; t[5] = b.y; t[6] = b.z; t[7] = b.w;
    return __builtin_bit_cast(v16bf, t);
}

__device__ __forceinline__ v8f wmma_bf16(v16bf a, v16bf b, v8f c) {
    return __builtin_amdgcn_wmma_f32_16x16x32_bf16(
        /*neg_a=*/false, a, /*neg_b=*/false, b,
        /*c_mod=*/(short)0, c, /*reuse_a=*/false, /*reuse_b=*/false);
}

// ---------------------------------------------------------------------------
// Pre-pass: convert the 12 512x512 f32 matrices (Uf/Uu/Uo1/Wo2 x 3 layers)
// into bf16 WMMA-B-fragment-major order in d_ws.
// dst dword index = mat*131072 + ((kt*32 + nt)*32 + lane)*8 + j
// lane half h covers K = kt*32 + 16h + [0,16); VGPR j packs K pair (2j, 2j+1).
// ---------------------------------------------------------------------------
__global__ void dgm_pack_weights(const float* __restrict__ Uf,
                                 const float* __restrict__ Uu,
                                 const float* __restrict__ Uo1,
                                 const float* __restrict__ Wo2,
                                 uint32_t* __restrict__ wsp)
{
    int gid = blockIdx.x * 256 + threadIdx.x;
    if (gid >= 12 * MAT_DW) return;
    int mat = gid >> 17;            // / 131072
    int rem = gid & (MAT_DW - 1);
    int j   = rem & 7;
    int L   = (rem >> 3) & 31;
    int nt  = (rem >> 8) & 31;
    int kt  = rem >> 13;            // 0..15
    int c   = L & 15, hh = L >> 4;
    int k0  = kt * 32 + hh * 16 + 2 * j;
    int n   = nt * 16 + c;
    int layer = mat >> 2, which = mat & 3;
    const float* src = (which == 0) ? Uf : (which == 1) ? Uu : (which == 2) ? Uo1 : Wo2;
    src += layer * (512 * 512);
    uint32_t lo = bf16_rne(src[k0 * 512 + n]);
    uint32_t hi = bf16_rne(src[(k0 + 1) * 512 + n]);
    wsp[gid] = lo | (hi << 16);
}

// ---------------------------------------------------------------------------
// GEMM micro-kernel: acc[n][s] += A(4 row strips from LDS) @ B(2 N-tiles).
// Each B fragment is reused by 4 WMMAs (the 4 row strips) -> minimal L2 reads.
// ---------------------------------------------------------------------------
__device__ __forceinline__ void gemm_2n4s(const uint32_t* __restrict__ smem, int srcDw,
                                          const uint32_t* __restrict__ frag,
                                          int nt0, int lane, int lh, int lc,
                                          v8f acc[2][4])
{
    for (int ks = 0; ks < 16; ++ks) {
        v16bf a[4];
#pragma unroll
        for (int s = 0; s < 4; ++s) {
            const int ao = srcDw + (s * 16 + lc) * 256 + ks * 16 + 8 * lh;
            a[s] = make_frag(*(const uint4*)&smem[ao], *(const uint4*)&smem[ao + 4]);
        }
        v16bf b[2];
#pragma unroll
        for (int n = 0; n < 2; ++n) {
            const int fb = ((ks * 32 + nt0 + n) * 32 + lane) * 8;
            b[n] = make_frag(*(const uint4*)(frag + fb), *(const uint4*)(frag + fb + 4));
        }
#pragma unroll
        for (int n = 0; n < 2; ++n)
#pragma unroll
            for (int s = 0; s < 4; ++s)
                acc[n][s] = wmma_bf16(a[s], b[n], acc[n][s]);
    }
}

// Accumulator init with the K=4 X@W term + bias, laid out per WMMA C layout:
// lane -> N = lc; VGPR v -> M = v + 8*lh (within each 16-row strip).
__device__ __forceinline__ void init_xw(const float* __restrict__ xs,
                                        const float* __restrict__ W,
                                        const float* __restrict__ bias,
                                        int nt0, int lh, int lc, v8f acc[2][4])
{
#pragma unroll
    for (int n = 0; n < 2; ++n) {
        const int ncol = (nt0 + n) * 16 + lc;
        const float bb = bias[ncol];
        const float w0 = W[0 * 512 + ncol], w1 = W[1 * 512 + ncol];
        const float w2 = W[2 * 512 + ncol], w3 = W[3 * 512 + ncol];
#pragma unroll
        for (int s = 0; s < 4; ++s)
#pragma unroll
            for (int v = 0; v < 8; ++v) {
                const int xr = s * 16 + v + 8 * lh;
                float a = fmaf(xs[xr * 4 + 0], w0, bb);
                a = fmaf(xs[xr * 4 + 1], w1, a);
                a = fmaf(xs[xr * 4 + 2], w2, a);
                a = fmaf(xs[xr * 4 + 3], w3, a);
                acc[n][s][v] = a;
            }
    }
}

// ---------------------------------------------------------------------------
// Main fused kernel: one block = 64 batch rows x full 512-node state.
// 8 waves partition N: wave w owns N-tiles 4w..4w+3 (two groups of 2),
// and computes all 4 row strips for each.
// ---------------------------------------------------------------------------
__global__ __launch_bounds__(256, 1)
void dgm_main(const float* __restrict__ X,    const float* __restrict__ W_in,
              const float* __restrict__ b_in,
              const float* __restrict__ Wf,   const float* __restrict__ bfv,
              const float* __restrict__ Wu,   const float* __restrict__ buv,
              const float* __restrict__ Wo1,  const float* __restrict__ bo1v,
              const float* __restrict__ bo2v,
              const float* __restrict__ W_out, const float* __restrict__ b_out,
              const uint32_t* __restrict__ wfrag,
              float* __restrict__ out)
{
    extern __shared__ uint32_t smem[];
    uint16_t* s16 = reinterpret_cast<uint16_t*>(smem);
    float* xs  = reinterpret_cast<float*>(smem + XS_DW);
    float* red = reinterpret_cast<float*>(smem + RED_DW);

    const int tid  = threadIdx.x;
    const int row0 = blockIdx.x * 64;

    // stage X tile (64 rows x 4)
    xs[tid] = X[row0 * 4 + tid];
    __syncthreads();

    // H = silu(X @ W_in + b_in)  -> bf16 in LDS (two cols per dword store)
    for (int e = tid * 2; e < 64 * 512; e += 512) {
        int row = e >> 9, col = e & 511;
        float s0 = b_in[col], s1 = b_in[col + 1];
#pragma unroll
        for (int k = 0; k < 4; ++k) {
            float xv = xs[row * 4 + k];
            s0 = fmaf(xv, W_in[k * 512 + col],     s0);
            s1 = fmaf(xv, W_in[k * 512 + col + 1], s1);
        }
        smem[H_DW + (e >> 1)] = bf16_rne(fsilu(s0)) | (bf16_rne(fsilu(s1)) << 16);
    }
    __syncthreads();

    const int lane = tid & 31;
    const int wv   = tid >> 5;      // 0..7: owns N-tiles 4wv .. 4wv+3
    const int lh   = lane >> 4;
    const int lc   = lane & 15;

    for (int layer = 0; layer < 3; ++layer) {
        const float* WfL = Wf  + layer * 2048;  const float* bfL = bfv  + layer * 512;
        const float* WuL = Wu  + layer * 2048;  const float* buL = buv  + layer * 512;
        const float* WoL = Wo1 + layer * 2048;  const float* boL = bo1v + layer * 512;
        const uint32_t* fragF  = wfrag + (layer * 4 + 0) * MAT_DW;
        const uint32_t* fragU  = wfrag + (layer * 4 + 1) * MAT_DW;
        const uint32_t* fragO  = wfrag + (layer * 4 + 2) * MAT_DW;
        const uint32_t* fragW2 = wfrag + (layer * 4 + 3) * MAT_DW;

        // ---- Pass F: F = sigmoid(X@Wf + H@Uf + bf) -> F region ----
        for (int g = 0; g < 2; ++g) {
            const int nt0 = wv * 4 + g * 2;
            v8f acc[2][4];
            init_xw(xs, WfL, bfL, nt0, lh, lc, acc);
            gemm_2n4s(smem, H_DW, fragF, nt0, lane, lh, lc, acc);
#pragma unroll
            for (int n = 0; n < 2; ++n)
#pragma unroll
                for (int s = 0; s < 4; ++s)
#pragma unroll
                    for (int v = 0; v < 8; ++v) {
                        const int idx = (s * 16 + v + 8 * lh) * 512 + (nt0 + n) * 16 + lc;
                        s16[F_E + idx] = (uint16_t)bf16_rne(fsigmoid(acc[n][s][v]));
                    }
        }

        // ---- Pass U: T <- sigmoid(X@Wu + H@Uu + bu) ----
        for (int g = 0; g < 2; ++g) {
            const int nt0 = wv * 4 + g * 2;
            v8f acc[2][4];
            init_xw(xs, WuL, buL, nt0, lh, lc, acc);
            gemm_2n4s(smem, H_DW, fragU, nt0, lane, lh, lc, acc);
#pragma unroll
            for (int n = 0; n < 2; ++n)
#pragma unroll
                for (int s = 0; s < 4; ++s)
#pragma unroll
                    for (int v = 0; v < 8; ++v) {
                        const int idx = (s * 16 + v + 8 * lh) * 512 + (nt0 + n) * 16 + lc;
                        s16[T_E + idx] = (uint16_t)bf16_rne(fsigmoid(acc[n][s][v]));
                    }
        }

        // ---- Pass O1: T <- T * tanh(X@Wo1 + H@Uo1 + bo1) ----
        // Safe without barrier: each T element is written (pass U) and
        // re-read here by the same lane of the same wave.
        for (int g = 0; g < 2; ++g) {
            const int nt0 = wv * 4 + g * 2;
            v8f acc[2][4];
            init_xw(xs, WoL, boL, nt0, lh, lc, acc);
            gemm_2n4s(smem, H_DW, fragO, nt0, lane, lh, lc, acc);
#pragma unroll
            for (int n = 0; n < 2; ++n)
#pragma unroll
                for (int s = 0; s < 4; ++s)
#pragma unroll
                    for (int v = 0; v < 8; ++v) {
                        const int idx = (s * 16 + v + 8 * lh) * 512 + (nt0 + n) * 16 + lc;
                        const float usig = bf16_to_f(s16[T_E + idx]);
                        s16[T_E + idx] =
                            (uint16_t)bf16_rne(usig * ftanh_(acc[n][s][v]));
                    }
        }
        __syncthreads();   // all of T visible to all waves (A operand of O2)

        // ---- Pass O2: O2 = silu(T@Wo2 + bo2);  H = F*H + O2 (in place) ----
        for (int g = 0; g < 2; ++g) {
            const int nt0 = wv * 4 + g * 2;
            v8f acc[2][4];
#pragma unroll
            for (int n = 0; n < 2; ++n) {
                const float b2 = bo2v[layer * 512 + (nt0 + n) * 16 + lc];
#pragma unroll
                for (int s = 0; s < 4; ++s)
#pragma unroll
                    for (int v = 0; v < 8; ++v) acc[n][s][v] = b2;
            }
            gemm_2n4s(smem, T_DW, fragW2, nt0, lane, lh, lc, acc);
#pragma unroll
            for (int n = 0; n < 2; ++n)
#pragma unroll
                for (int s = 0; s < 4; ++s)
#pragma unroll
                    for (int v = 0; v < 8; ++v) {
                        const int idx = (s * 16 + v + 8 * lh) * 512 + (nt0 + n) * 16 + lc;
                        const float o2 = fsilu(acc[n][s][v]);
                        const float fg = bf16_to_f(s16[F_E + idx]);
                        const float hv = bf16_to_f(s16[idx]);
                        s16[idx] = (uint16_t)bf16_rne(fmaf(fg, hv, o2));
                    }
        }
        __syncthreads();   // H update visible before next layer reads it
    }

    // ---- out = silu(H @ W_out + b_out): 4 threads per row ----
    {
        int row = tid >> 2, part = tid & 3;
        float s = 0.0f;
        for (int c2 = part * 128; c2 < part * 128 + 128; ++c2)
            s = fmaf(bf16_to_f(s16[row * 512 + c2]), W_out[c2], s);
        red[tid] = s;
        __syncthreads();
        if (part == 0) {
            float tot = red[tid] + red[tid + 1] + red[tid + 2] + red[tid + 3] + b_out[0];
            out[row0 + row] = fsilu(tot);
        }
    }
}

extern "C" void kernel_launch(void* const* d_in, const int* in_sizes, int n_in,
                              void* d_out, int out_size, void* d_ws, size_t ws_size,
                              hipStream_t stream) {
    (void)in_sizes; (void)n_in; (void)out_size; (void)ws_size;
    const float* X     = (const float*)d_in[0];
    const float* W_in  = (const float*)d_in[1];
    const float* b_in  = (const float*)d_in[2];
    const float* Wf    = (const float*)d_in[3];
    const float* Uf    = (const float*)d_in[4];
    const float* bf_   = (const float*)d_in[5];
    const float* Wu    = (const float*)d_in[6];
    const float* Uu    = (const float*)d_in[7];
    const float* bu_   = (const float*)d_in[8];
    const float* Wo1   = (const float*)d_in[9];
    const float* Uo1   = (const float*)d_in[10];
    const float* bo1_  = (const float*)d_in[11];
    const float* Wo2   = (const float*)d_in[12];
    const float* bo2_  = (const float*)d_in[13];
    const float* W_out = (const float*)d_in[14];
    const float* b_out = (const float*)d_in[15];
    uint32_t* wsp = (uint32_t*)d_ws;
    float* out = (float*)d_out;

    // 12 * 131072 packed dwords
    dgm_pack_weights<<<6144, 256, 0, stream>>>(Uf, Uu, Uo1, Wo2, wsp);

    (void)hipFuncSetAttribute((const void*)dgm_main,
                              hipFuncAttributeMaxDynamicSharedMemorySize,
                              (int)SMEM_BYTES);
    dgm_main<<<131072 / 64, 256, SMEM_BYTES, stream>>>(
        X, W_in, b_in, Wf, bf_, Wu, bu_, Wo1, bo1_, bo2_, W_out, b_out, wsp, out);
}
